// ModelSeA_58858231825040
// MI455X (gfx1250) — matrix operand order repaired and checked by reading, not compiled, not run on hardware
//
#include <hip/hip_runtime.h>

typedef __attribute__((ext_vector_type(16))) __bf16 v16bf;
typedef __attribute__((ext_vector_type(8)))  float  v8f;
typedef __attribute__((ext_vector_type(4)))  float  v4f;

#define T_ 2
#define P_ 32768
#define D_ 1600
#define H_ 64
#define N_ATOMS (T_ * P_)
#define KB0 (D_ / 32)          // 50 k-steps for layer 0
#define KB1 (H_ / 32)          // 2 k-steps for layers 1/2
#define FRAG_ELEMS (32 * 16)   // one B fragment: 32 lanes x 16 bf16
#define W0P_ELEMS (T_ * KB0 * 4 * FRAG_ELEMS)   // 204800
#define W1P_ELEMS (T_ * KB1 * 4 * FRAG_ELEMS)   // 8192
#define WAVES_PER_WG 8
#define ATOMS_PER_WAVE 32      // two 16-row M-tiles share each B fragment
#define HROW 68                // padded LDS row stride (floats): 16B-aligned, conflict-free

// ---------------- prep: pack weights into WMMA B-fragment order (bf16) ----------------
// B fragment layout (mirrors documented sparse 64x16 B layout, scaled to K=32):
//   lane L holds column n = nb*16 + (L&15); element j holds K = kb*32 + 16*(j>=8) + 8*(L>=16) + (j&7).
__global__ __launch_bounds__(256) void prep_kernel(
    const float* __restrict__ W0, const float* __restrict__ W1, const float* __restrict__ W2,
    __bf16* __restrict__ w0p, __bf16* __restrict__ w1p, __bf16* __restrict__ w2p,
    float* __restrict__ out)
{
    int idx = blockIdx.x * 256 + threadIdx.x;
    if (idx == 0) out[0] = 0.0f;     // zero total-energy accumulator every call

    if (idx < W0P_ELEMS) {
        int j  = idx & 15;
        int L  = (idx >> 4) & 31;
        int nb = (idx >> 9) & 3;
        int tk = idx >> 11;          // t*KB0 + kb
        int kb = tk % KB0;
        int t  = tk / KB0;
        int k  = kb * 32 + ((j & 8) << 1) + ((L >= 16) ? 8 : 0) + (j & 7);
        int n  = nb * 16 + (L & 15);
        w0p[idx] = (__bf16)W0[((size_t)t * D_ + k) * H_ + n];
    } else if (idx < W0P_ELEMS + 2 * W1P_ELEMS) {
        int i2 = idx - W0P_ELEMS;
        const float* Wsrc = (i2 < W1P_ELEMS) ? W1 : W2;
        __bf16* dst = (i2 < W1P_ELEMS) ? w1p : w2p;
        int i = (i2 < W1P_ELEMS) ? i2 : (i2 - W1P_ELEMS);
        int j  = i & 15;
        int L  = (i >> 4) & 31;
        int nb = (i >> 9) & 3;
        int kb = (i >> 11) & (KB1 - 1);
        int t  = i >> 12;
        int k  = kb * 32 + ((j & 8) << 1) + ((L >= 16) ? 8 : 0) + (j & 7);
        int n  = nb * 16 + (L & 15);
        dst[i] = (__bf16)Wsrc[((size_t)t * H_ + k) * H_ + n];
    }
}

// ---------------- device helpers ----------------
__device__ __forceinline__ float fast_tanh(float x) {
    // tanh(x) = (e-1)/(e+1), e = exp2(2*log2(e)*x); clamp avoids inf/inf -> NaN
    x = fminf(fmaxf(x, -15.0f), 15.0f);
    float e = __builtin_amdgcn_exp2f(x * 2.885390081777927f);
    return (e - 1.0f) * __builtin_amdgcn_rcpf(e + 1.0f);
}

// Build 16x32 bf16 A fragment from two 8-float runs (ISA layout §7.12.2):
// lane<16 : VGPR0-3 = K0..7 (pairs), VGPR4-7 = K16..23 ; lane>=16: +8 on both runs.
__device__ __forceinline__ v16bf cvt_frag(v4f f0, v4f f1, v4f f2, v4f f3) {
    v16bf a;
    a[0]  = (__bf16)f0[0]; a[1]  = (__bf16)f0[1]; a[2]  = (__bf16)f0[2]; a[3]  = (__bf16)f0[3];
    a[4]  = (__bf16)f1[0]; a[5]  = (__bf16)f1[1]; a[6]  = (__bf16)f1[2]; a[7]  = (__bf16)f1[3];
    a[8]  = (__bf16)f2[0]; a[9]  = (__bf16)f2[1]; a[10] = (__bf16)f2[2]; a[11] = (__bf16)f2[3];
    a[12] = (__bf16)f3[0]; a[13] = (__bf16)f3[1]; a[14] = (__bf16)f3[2]; a[15] = (__bf16)f3[3];
    return a;
}

__device__ __forceinline__ v16bf load_a_frag_nt(const float* row, int k0) {
    v4f f0 = __builtin_nontemporal_load((const v4f*)(row + k0));
    v4f f1 = __builtin_nontemporal_load((const v4f*)(row + k0 + 4));
    v4f f2 = __builtin_nontemporal_load((const v4f*)(row + k0 + 16));
    v4f f3 = __builtin_nontemporal_load((const v4f*)(row + k0 + 20));
    return cvt_frag(f0, f1, f2, f3);
}

// ---------------- main fused kernel: one wave per 32 atoms (two 16-row M-tiles) ----------------
__global__ __launch_bounds__(256) void fit_kernel(
    const float* __restrict__ descrpt,
    const float* __restrict__ b0v,  const float* __restrict__ b1v,
    const float* __restrict__ idt1, const float* __restrict__ b2v,
    const float* __restrict__ idt2, const float* __restrict__ Wf,
    const float* __restrict__ bfv,  const float* __restrict__ biasv,
    const __bf16* __restrict__ w0p, const __bf16* __restrict__ w1p,
    const __bf16* __restrict__ w2p, float* __restrict__ out)
{
    __shared__ float h_lds[WAVES_PER_WG][16 * HROW];
    __shared__ float e_acc;

    const int tid  = threadIdx.x;
    const int wave = tid >> 5;
    const int lane = tid & 31;
    const int hh   = lane >> 4;   // half: 0 or 1
    const int col  = lane & 15;

    if (tid == 0) e_acc = 0.0f;
    __syncthreads();

    const int mbase = (blockIdx.x * WAVES_PER_WG + wave) * ATOMS_PER_WAVE;
    const int t     = (mbase >= P_) ? 1 : 0;   // atom type (tiles never straddle boundary)

    // ===== layer 0: [32 x 1600] x [1600 x 64], B fragments reused by both M-tiles =====
    v8f acc0[4] = {v8f{}, v8f{}, v8f{}, v8f{}};
    v8f acc1[4] = {v8f{}, v8f{}, v8f{}, v8f{}};
    const float* arow0 = descrpt + (size_t)(mbase + col) * D_;
    const float* arow1 = descrpt + (size_t)(mbase + 16 + col) * D_;
    const __bf16* w0t  = w0p + (size_t)t * (KB0 * 4 * FRAG_ELEMS);

    #pragma unroll 2
    for (int kb = 0; kb < KB0; ++kb) {
        const int k0 = kb * 32 + hh * 8;
        v16bf a0 = load_a_frag_nt(arow0, k0);
        v16bf a1 = load_a_frag_nt(arow1, k0);
        const __bf16* wkb = w0t + kb * (4 * FRAG_ELEMS);
        #pragma unroll
        for (int nb = 0; nb < 4; ++nb) {
            v16bf b = *(const v16bf*)(wkb + nb * FRAG_ELEMS + lane * 16);
            acc0[nb] = __builtin_amdgcn_wmma_f32_16x16x32_bf16(
                false, a0, false, b, (short)0, acc0[nb], false, false);
            acc1[nb] = __builtin_amdgcn_wmma_f32_16x16x32_bf16(
                false, a1, false, b, (short)0, acc1[nb], false, false);
        }
    }

    // per-lane column constants
    float c_b0[4], c_b1[4], c_i1[4], c_b2[4], c_i2[4], wfv[4];
    #pragma unroll
    for (int nb = 0; nb < 4; ++nb) {
        const int n = t * H_ + nb * 16 + col;
        c_b0[nb] = b0v[n];  c_b1[nb] = b1v[n];  c_i1[nb] = idt1[n];
        c_b2[nb] = b2v[n];  c_i2[nb] = idt2[n]; wfv[nb]  = Wf[n];
    }
    const float aeb = bfv[t] + biasv[t];
    const __bf16* w1t = w1p + (size_t)t * (KB1 * 4 * FRAG_ELEMS);
    const __bf16* w2t = w2p + (size_t)t * (KB1 * 4 * FRAG_ELEMS);
    const float* brow = &h_lds[wave][col * HROW];

    float es = 0.0f;   // this lane's contribution to total energy

    // ===== per-tile: bias+tanh, two resnet layers (LDS bounce), final dot =====
    #pragma unroll
    for (int tile = 0; tile < 2; ++tile) {
        v8f* acc = tile ? acc1 : acc0;

        #pragma unroll
        for (int nb = 0; nb < 4; ++nb)
            #pragma unroll
            for (int r = 0; r < 8; ++r)
                acc[nb][r] = fast_tanh(acc[nb][r] + c_b0[nb]);

        #pragma unroll
        for (int layer = 0; layer < 2; ++layer) {
            const __bf16* wlt = layer ? w2t : w1t;
            const float*  bl  = layer ? c_b2 : c_b1;
            const float*  il  = layer ? c_i2 : c_i1;

            // stage h tile (C layout) into LDS, re-read in A-fragment layout
            #pragma unroll
            for (int nb = 0; nb < 4; ++nb)
                #pragma unroll
                for (int r = 0; r < 8; ++r)
                    h_lds[wave][(hh * 8 + r) * HROW + nb * 16 + col] = acc[nb][r];

            v8f upd[4] = {v8f{}, v8f{}, v8f{}, v8f{}};
            #pragma unroll
            for (int kb = 0; kb < KB1; ++kb) {
                const int k0 = kb * 32 + hh * 8;
                v4f f0 = *(const v4f*)(brow + k0);
                v4f f1 = *(const v4f*)(brow + k0 + 4);
                v4f f2 = *(const v4f*)(brow + k0 + 16);
                v4f f3 = *(const v4f*)(brow + k0 + 20);
                v16bf a = cvt_frag(f0, f1, f2, f3);
                const __bf16* wkb = wlt + kb * (4 * FRAG_ELEMS);
                #pragma unroll
                for (int nb = 0; nb < 4; ++nb) {
                    v16bf b = *(const v16bf*)(wkb + nb * FRAG_ELEMS + lane * 16);
                    upd[nb] = __builtin_amdgcn_wmma_f32_16x16x32_bf16(
                        false, a, false, b, (short)0, upd[nb], false, false);
                }
            }
            #pragma unroll
            for (int nb = 0; nb < 4; ++nb)
                #pragma unroll
                for (int r = 0; r < 8; ++r)
                    acc[nb][r] += fast_tanh(upd[nb][r] + bl[nb]) * il[nb];
        }

        // final: ae[m] = sum_n h[m][n]*Wf[t][n] + bf[t] + bias[t]
        float p[8];
        #pragma unroll
        for (int r = 0; r < 8; ++r)
            p[r] = acc[0][r] * wfv[0] + acc[1][r] * wfv[1]
                 + acc[2][r] * wfv[2] + acc[3][r] * wfv[3];
        #pragma unroll
        for (int off = 1; off < 16; off <<= 1)
            #pragma unroll
            for (int r = 0; r < 8; ++r) p[r] += __shfl_xor(p[r], off, 32);

        if (col == 0) {
            #pragma unroll
            for (int r = 0; r < 8; ++r) {
                float ae = p[r] + aeb;
                out[1 + mbase + tile * 16 + hh * 8 + r] = ae;   // energy_raw
                es += ae;
            }
        }
    }

    if (col == 0) atomicAdd(&e_acc, es);
    __syncthreads();
    if (tid == 0) atomicAdd(&out[0], e_acc);    // total energy
}

// ---------------- host launcher ----------------
extern "C" void kernel_launch(void* const* d_in, const int* in_sizes, int n_in,
                              void* d_out, int out_size, void* d_ws, size_t ws_size,
                              hipStream_t stream) {
    const float* descrpt = (const float*)d_in[0];
    const float* W0   = (const float*)d_in[1];
    const float* b0   = (const float*)d_in[2];
    const float* W1   = (const float*)d_in[3];
    const float* b1   = (const float*)d_in[4];
    const float* idt1 = (const float*)d_in[5];
    const float* W2   = (const float*)d_in[6];
    const float* b2   = (const float*)d_in[7];
    const float* idt2 = (const float*)d_in[8];
    const float* Wf   = (const float*)d_in[9];
    const float* bf   = (const float*)d_in[10];
    const float* bias = (const float*)d_in[11];
    float* out = (float*)d_out;

    __bf16* w0p = (__bf16*)d_ws;
    __bf16* w1p = (__bf16*)((char*)d_ws + (size_t)W0P_ELEMS * 2);
    __bf16* w2p = (__bf16*)((char*)d_ws + (size_t)(W0P_ELEMS + W1P_ELEMS) * 2);

    const int prep_elems  = W0P_ELEMS + 2 * W1P_ELEMS;
    const int prep_blocks = (prep_elems + 255) / 256;
    prep_kernel<<<prep_blocks, 256, 0, stream>>>(W0, W1, W2, w0p, w1p, w2p, out);

    const int blocks = N_ATOMS / (ATOMS_PER_WAVE * WAVES_PER_WG);   // 256
    fit_kernel<<<blocks, 256, 0, stream>>>(descrpt, b0, b1, idt1, b2, idt2,
                                           Wf, bf, bias, w0p, w1p, w2p, out);
}
